// SpatialAttention_15994458211365
// MI455X (gfx1250) — matrix operand order repaired and checked by reading, not compiled, not run on hardware
//
#include <hip/hip_runtime.h>
#include <math.h>

// ---------------------------------------------------------------------------
// Fused FiLM-conditioned spatial attention for MI455X (gfx1250, wave32, WMMA)
// B=16, T=512, J=27, D=256, H=4, dk=64.  One block per (b,t); bf16 WMMA with
// fp32 accumulation; softmax in fp32.
// ---------------------------------------------------------------------------

#define Bn   16
#define Tn   512
#define Jn   27
#define Dn   256
#define Sn   64
#define Hn   4

typedef __attribute__((ext_vector_type(16))) __bf16 v16bf;
typedef __attribute__((ext_vector_type(8)))  float  v8f;

// Native bf16 convert (RNE); lowers to the hardware cvt on gfx1250.
__device__ __forceinline__ unsigned short f2bf(float f) {
    union { __bf16 h; unsigned short u; } c;
    c.h = (__bf16)f;
    return c.u;
}

__device__ __forceinline__ unsigned pack2(float a, float b) {
    return (unsigned)f2bf(a) | ((unsigned)f2bf(b) << 16);
}

// 16-bit A-matrix (16xK=32) element (m,kk) -> lane*16 + vgpr-half index.
// ISA 7.12.2: lanes 0-15 hold K 0-7 (VGPR0-3), lanes 16-31 hold K 8-15;
// K 16-31 repeat the pattern in VGPRs 4-7.
__device__ __forceinline__ int amap(int m, int kk) {
    int khi = kk >> 4, klo = kk & 15;
    int lane = m + ((klo >> 3) << 4);
    int e    = (khi << 3) + (klo & 7);
    return lane * 16 + e;
}

// 16-bit B-matrix (K=32 x 16) element (kk,n): same K order as amap —
// lanes 0-15 hold K 0-7,16-23; lanes 16-31 hold K 8-15,24-31.
__device__ __forceinline__ int bmap(int kk, int n) {
    int lane = n + (((kk >> 3) & 1) << 4);
    int e    = ((kk >> 4) << 3) + (kk & 7);
    return lane * 16 + e;
}

// ---------------------------------------------------------------------------
// Pre-pass 1: convert the four 256x256 fp32 weights into bf16 B-fragments.
// ---------------------------------------------------------------------------
__global__ __launch_bounds__(256) void prep_weights_kernel(
    const float* __restrict__ Wq, const float* __restrict__ Wk,
    const float* __restrict__ Wv, const float* __restrict__ Wo,
    unsigned short* __restrict__ wt)
{
    int widx = blockIdx.y;
    const float* src = (widx == 0) ? Wq : (widx == 1) ? Wk : (widx == 2) ? Wv : Wo;
    unsigned short* dst = wt + (size_t)widx * 65536;
    int n = blockIdx.x;        // output dim (row of W)
    int k = threadIdx.x;       // input dim  (col of W)
    int nt = n >> 4, nl = n & 15, kt = k >> 5, kk = k & 31;
    dst[(kt * 16 + nt) * 512 + bmap(kk, nl)] = f2bf(src[n * Dn + k]);
}

// ---------------------------------------------------------------------------
// Pre-pass 2: FiLM scale/shift per (b, d).
// ---------------------------------------------------------------------------
__global__ __launch_bounds__(256) void film_kernel(
    const float* __restrict__ se,  const float* __restrict__ Ws,
    const float* __restrict__ bs,  const float* __restrict__ Wsh,
    const float* __restrict__ bsh, float* __restrict__ scale,
    float* __restrict__ shift)
{
    int b = blockIdx.x, d = threadIdx.x;
    float a0 = bs[d], a1 = bsh[d];
    #pragma unroll 8
    for (int s = 0; s < Sn; ++s) {
        float e = se[b * Sn + s];
        a0 = fmaf(e, Ws[d * Sn + s], a0);
        a1 = fmaf(e, Wsh[d * Sn + s], a1);
    }
    scale[b * Dn + d] = tanhf(a0);
    shift[b * Dn + d] = a1;
}

// ---------------------------------------------------------------------------
// Main fused kernel: one block per (b,t).
// ---------------------------------------------------------------------------
__global__ __launch_bounds__(256) void fused_spatial_attn_kernel(
    const float* __restrict__ h,
    const unsigned short* __restrict__ Wq_t,
    const unsigned short* __restrict__ Wk_t,
    const unsigned short* __restrict__ Wv_t,
    const unsigned short* __restrict__ Wo_t,
    const float* __restrict__ film_scale,
    const float* __restrict__ film_shift,
    float* __restrict__ out)
{
    // 64 KB LDS total, with stage-wise reuse:
    __shared__ __attribute__((aligned(64))) float          buf0f[4096]; // hA | scores | outA
    __shared__ __attribute__((aligned(64))) unsigned short buf1[8192];  // qA | probsA
    __shared__ __attribute__((aligned(64))) unsigned short buf2[8192];  // kB
    __shared__ __attribute__((aligned(64))) unsigned short buf3[8192];  // vB
    unsigned short* buf0h = (unsigned short*)buf0f;

    const int bt    = blockIdx.x;
    const int b     = bt / Tn;
    const int tid   = threadIdx.x;
    const int lane  = tid & 31;
    const int wv    = tid >> 5;          // wave id 0..7
    const int ln16  = lane & 15;
    const int lhalf = lane >> 4;
    const int m0    = lhalf << 3;        // base row-in-tile of this lane's accs

    const float* hb = h + (size_t)bt * Jn * Dn;

    // Preload FiLM params for this thread's two output columns.
    const int ncol0 = (wv * 2) * 16 + ln16;
    const int ncol1 = ncol0 + 16;
    const float fsc0 = 1.0f + film_scale[b * Dn + ncol0];
    const float fsc1 = 1.0f + film_scale[b * Dn + ncol1];
    const float fsh0 = film_shift[b * Dn + ncol0];
    const float fsh1 = film_shift[b * Dn + ncol1];

    // ---- stage 0: h tile (pad rows 27..31 with zeros) -> bf16 A-frags -----
    // 8 consecutive k (k8-aligned) land in 8 contiguous bf16 slots of one
    // fragment -> two b128 loads + one b128 LDS store per group.
    #pragma unroll
    for (int it = 0; it < 4; ++it) {
        int g  = tid + it * 256;          // 8-float group, 1024 total
        int j  = g >> 5;                  // row 0..31
        int k8 = (g & 31) << 3;           // col base, multiple of 8
        float4 va = make_float4(0.f, 0.f, 0.f, 0.f);
        float4 vb = make_float4(0.f, 0.f, 0.f, 0.f);
        if (j < Jn) {
            va = *(const float4*)&hb[j * Dn + k8];
            vb = *(const float4*)&hb[j * Dn + k8 + 4];
        }
        int mt = j >> 4, m = j & 15, kt = k8 >> 5;
        uint4 pk;
        pk.x = pack2(va.x, va.y);
        pk.y = pack2(va.z, va.w);
        pk.z = pack2(vb.x, vb.y);
        pk.w = pack2(vb.z, vb.w);
        *(uint4*)&buf0h[(mt * 8 + kt) * 512 + amap(m, k8 & 31)] = pk;
    }
    __syncthreads();

    // ---- stage 1: q,k,v projections; FiLM on k,v; scatter to frag layouts -
    #pragma unroll
    for (int p = 0; p < 3; ++p) {
        const unsigned short* Wt = (p == 0) ? Wq_t : (p == 1) ? Wk_t : Wv_t;
        #pragma unroll
        for (int mt = 0; mt < 2; ++mt) {
            #pragma unroll
            for (int nti = 0; nti < 2; ++nti) {
                int nt = wv * 2 + nti;
                v8f acc = {};
                #pragma unroll
                for (int kt = 0; kt < 8; ++kt) {
                    v16bf a  = *(const v16bf*)&buf0h[(mt * 8 + kt) * 512 + lane * 16];
                    v16bf bm = *(const v16bf*)&Wt[(size_t)(kt * 16 + nt) * 512 + lane * 16];
                    acc = __builtin_amdgcn_wmma_f32_16x16x32_bf16(
                        false, a, false, bm, (short)0, acc, false, false);
                }
                int n = nt * 16 + ln16;
                int head = n >> 6, kd = n & 63;
                float sc = (nti == 0 ? fsc0 : fsc1);
                float sh = (nti == 0 ? fsh0 : fsh1);
                float fv[8];
                #pragma unroll
                for (int r = 0; r < 8; ++r)
                    fv[r] = (p == 0) ? acc[r] : fmaf(acc[r], sc, sh);
                if (p == 0) {
                    // q: A-frags [head][mt][ktA]; addresses linear in r
                    unsigned short* dst =
                        &buf1[((head * 2 + mt) * 2 + (kd >> 5)) * 512 + amap(m0, kd & 31)];
                    #pragma unroll
                    for (int r = 0; r < 8; ++r) dst[r * 16] = f2bf(fv[r]);
                } else if (p == 1) {
                    // k^T: B-frags [head][ktB][ntB]; addresses linear in r
                    unsigned short* dst =
                        &buf2[((head * 2 + (kd >> 5)) * 2 + mt) * 512 + bmap(kd & 31, m0)];
                    #pragma unroll
                    for (int r = 0; r < 8; ++r) dst[r * 16] = f2bf(fv[r]);
                } else {
                    // v: B-frags [n>>4]; 8 contiguous half-words -> one b128
                    uint4 pk;
                    pk.x = pack2(fv[0], fv[1]);
                    pk.y = pack2(fv[2], fv[3]);
                    pk.z = pack2(fv[4], fv[5]);
                    pk.w = pack2(fv[6], fv[7]);
                    *(uint4*)&buf3[(n >> 4) * 512 + (ln16 + 16 * lhalf) * 16 + 8 * mt] = pk;
                }
            }
        }
    }
    __syncthreads();

    // ---- stage 2: scores = q @ k^T * 1/sqrt(dk) --------------------------
    #pragma unroll
    for (int i2 = 0; i2 < 2; ++i2) {
        int tt = wv * 2 + i2;                    // 16 tiles
        int head = tt >> 2, mt = (tt >> 1) & 1, nt = tt & 1;
        v8f acc = {};
        #pragma unroll
        for (int kt = 0; kt < 2; ++kt) {
            v16bf a  = *(const v16bf*)&buf1[((head * 2 + mt) * 2 + kt) * 512 + lane * 16];
            v16bf bm = *(const v16bf*)&buf2[((head * 2 + kt) * 2 + nt) * 512 + lane * 16];
            acc = __builtin_amdgcn_wmma_f32_16x16x32_bf16(
                false, a, false, bm, (short)0, acc, false, false);
        }
        float* srow = &buf0f[head * 1024 + (mt * 16 + m0) * 32 + nt * 16 + ln16];
        #pragma unroll
        for (int r = 0; r < 8; ++r) srow[r * 32] = acc[r] * 0.125f;  // 1/sqrt(64)
    }
    __syncthreads();

    // ---- stage 3: masked softmax over the 27 real joints -----------------
    if (tid < 128) {
        int head = tid >> 5, row = tid & 31;
        float pr[32];
        if (row < Jn) {
            const float* srow = buf0f + head * 1024 + row * 32;
            float mx = -1e30f;
            #pragma unroll
            for (int c = 0; c < Jn; ++c) mx = fmaxf(mx, srow[c]);
            float sum = 0.0f;
            #pragma unroll
            for (int c = 0; c < Jn; ++c) { float e = __expf(srow[c] - mx); pr[c] = e; sum += e; }
            float inv = 1.0f / sum;
            #pragma unroll
            for (int c = 0; c < Jn; ++c) pr[c] *= inv;
            #pragma unroll
            for (int c = Jn; c < 32; ++c) pr[c] = 0.0f;
        } else {
            #pragma unroll
            for (int c = 0; c < 32; ++c) pr[c] = 0.0f;
        }
        // probs -> A-frag layout: lane m holds c{0..7,16..23}, lane m+16 holds
        // c{8..15,24..31}; each run of 8 is contiguous -> 4 b128 stores.
        int mt = row >> 4, m = row & 15;
        unsigned short* base = &buf1[(head * 2 + mt) * 512 + m * 16];
        uint4 pa, pb;
        pa.x = pack2(pr[0],  pr[1]);  pa.y = pack2(pr[2],  pr[3]);
        pa.z = pack2(pr[4],  pr[5]);  pa.w = pack2(pr[6],  pr[7]);
        pb.x = pack2(pr[16], pr[17]); pb.y = pack2(pr[18], pr[19]);
        pb.z = pack2(pr[20], pr[21]); pb.w = pack2(pr[22], pr[23]);
        *(uint4*)&base[0] = pa;
        *(uint4*)&base[8] = pb;
        pa.x = pack2(pr[8],  pr[9]);  pa.y = pack2(pr[10], pr[11]);
        pa.z = pack2(pr[12], pr[13]); pa.w = pack2(pr[14], pr[15]);
        pb.x = pack2(pr[24], pr[25]); pb.y = pack2(pr[26], pr[27]);
        pb.z = pack2(pr[28], pr[29]); pb.w = pack2(pr[30], pr[31]);
        *(uint4*)&base[256] = pa;     // lane m+16 -> +16*16 half-words
        *(uint4*)&base[264] = pb;
    }
    __syncthreads();

    // ---- stage 4: out_head = probs @ v (single K=32 step) ----------------
    #pragma unroll
    for (int i4 = 0; i4 < 4; ++i4) {
        int tt = wv * 4 + i4;                    // 32 tiles
        int head = tt >> 3, mt = (tt >> 2) & 1, ntl = tt & 3;
        v16bf a  = *(const v16bf*)&buf1[(head * 2 + mt) * 512 + lane * 16];
        v16bf bm = *(const v16bf*)&buf3[(head * 4 + ntl) * 512 + lane * 16];
        v8f acc = {};
        acc = __builtin_amdgcn_wmma_f32_16x16x32_bf16(
            false, a, false, bm, (short)0, acc, false, false);
        int n = head * 64 + ntl * 16 + ln16;
        unsigned short* dst = &buf0h[(mt * 8 + (n >> 5)) * 512 + amap(m0, n & 31)];
        #pragma unroll
        for (int r = 0; r < 8; ++r) dst[r * 16] = f2bf(acc[r]);
    }
    __syncthreads();

    // ---- stage 5: final projection out @ Wo.T; store 27 real rows --------
    float* ob = out + (size_t)bt * Jn * Dn;
    #pragma unroll
    for (int mt = 0; mt < 2; ++mt) {
        #pragma unroll
        for (int nti = 0; nti < 2; ++nti) {
            int nt = wv * 2 + nti;
            v8f acc = {};
            #pragma unroll
            for (int kt = 0; kt < 8; ++kt) {
                v16bf a  = *(const v16bf*)&buf0h[(mt * 8 + kt) * 512 + lane * 16];
                v16bf bm = *(const v16bf*)&Wo_t[(size_t)(kt * 16 + nt) * 512 + lane * 16];
                acc = __builtin_amdgcn_wmma_f32_16x16x32_bf16(
                    false, a, false, bm, (short)0, acc, false, false);
            }
            int n = nt * 16 + ln16;
            #pragma unroll
            for (int r = 0; r < 8; ++r) {
                int j = mt * 16 + m0 + r;
                if (j < Jn) ob[j * Dn + n] = acc[r];
            }
        }
    }
}

// ---------------------------------------------------------------------------
extern "C" void kernel_launch(void* const* d_in, const int* in_sizes, int n_in,
                              void* d_out, int out_size, void* d_ws, size_t ws_size,
                              hipStream_t stream) {
    (void)in_sizes; (void)n_in; (void)out_size; (void)ws_size;
    const float* h   = (const float*)d_in[0];
    const float* se  = (const float*)d_in[1];
    const float* Ws  = (const float*)d_in[2];
    const float* bs  = (const float*)d_in[3];
    const float* Wsh = (const float*)d_in[4];
    const float* bsh = (const float*)d_in[5];
    const float* Wq  = (const float*)d_in[6];
    const float* Wk  = (const float*)d_in[7];
    const float* Wv  = (const float*)d_in[8];
    const float* Wo  = (const float*)d_in[9];

    // workspace layout: 4x 65536 bf16 weight frags (512 KB) + FiLM scale/shift
    unsigned short* wt = (unsigned short*)d_ws;
    float* film  = (float*)((char*)d_ws + 4u * 65536u * sizeof(unsigned short));
    float* scale = film;
    float* shift = film + Bn * Dn;

    prep_weights_kernel<<<dim3(256, 4), 256, 0, stream>>>(Wq, Wk, Wv, Wo, wt);
    film_kernel<<<Bn, 256, 0, stream>>>(se, Ws, bs, Wsh, bsh, scale, shift);
    fused_spatial_attn_kernel<<<Bn * Tn, 256, 0, stream>>>(
        h, wt, wt + 65536, wt + 131072, wt + 196608, scale, shift, (float*)d_out);
}
